// Model_64536178589723
// MI455X (gfx1250) — matrix operand order repaired and checked by reading, not compiled, not run on hardware
//
#include <hip/hip_runtime.h>

// ---------------------------------------------------------------------------
// FreTS-style model, gfx1250 (CDNA5) implementation.
// Math path: f16 operands, f32 accumulation via v_wmma_f32_16x16x32_f16.
// Wave tile: 32(M) x 64(N); double-buffered fragment pipeline so loads for
// iteration k+1 are in flight while iteration k's 8 WMMAs execute.
// ---------------------------------------------------------------------------

constexpr int CB   = 8;     // batch
constexpr int CT   = 512;   // seq len
constexpr int CN   = 64;    // enc_in
constexpr int CE   = 128;   // embed
constexpr int CH   = 256;   // hidden
constexpr int CP   = 96;    // pred len
constexpr int CF   = 257;   // T/2+1
constexpr int CF2  = 514;   // real+imag bins
constexpr int CF2P = 544;   // padded to multiple of 32
constexpr int CR   = 512;   // B*N independent series
constexpr int CSPLIT = 32;  // split-K factor for fc1
constexpr float CLAMBDA = 0.001f;

typedef __attribute__((ext_vector_type(16))) _Float16 v16h;
typedef __attribute__((ext_vector_type(8)))  _Float16 v8h;
typedef __attribute__((ext_vector_type(8)))  float    v8f;

union V16H { v16h v; v8h h[2]; _Float16 e[16]; };

__device__ __forceinline__ float leakyf(float x){ return x >= 0.0f ? x : 0.01f * x; }
__device__ __forceinline__ float sshrink(float y){
  return y > CLAMBDA ? y - CLAMBDA : (y < -CLAMBDA ? y + CLAMBDA : 0.0f);
}

__device__ __forceinline__ v8f wmma16(v16h a, v16h b, v8f c){
  // D = A(16x32 f16) x B(32x16 f16) + C(16x16 f32)
  return __builtin_amdgcn_wmma_f32_16x16x32_f16(false, a, false, b, (short)0, c, false, false);
}

__device__ __forceinline__ v16h zero_frag(){
  V16H r;
  #pragma unroll
  for (int i = 0; i < 16; ++i) r.e[i] = (_Float16)0.0f;
  return r.v;
}

// A fragment: 16-bit A 16x32 ISA layout.
// lanes 0-15 (half=0): M=lane, halfs 0..7 -> K=k0+0..7,  halfs 8..15 -> K=k0+16..23
// lanes16-31 (half=1): M=lane-16, halfs 0..7 -> K=k0+8..15, halfs 8..15 -> K=k0+24..31
__device__ __forceinline__ v16h load_a_frag(const _Float16* A, int lda, int m0,
                                            int k0, int half, int lm){
  const _Float16* p = A + (size_t)(m0 + lm) * lda + k0 + 8 * half;
  V16H r;
  r.h[0] = *(const v8h*)p;
  r.h[1] = *(const v8h*)(p + 16);
  return r.v;
}

// B fragment: 16-bit B 32x16 ISA layout; B stored transposed Bt[N][K] so the
// K values per lane are two contiguous 16-byte loads.
// lanes 0-15: N=lane, halfs 0..7 -> K=k0+0..7, 8..15 -> K=k0+16..23 ; lanes16-31: N=lane-16, K=k0+8..15, k0+24..31
__device__ __forceinline__ v16h load_b_frag(const _Float16* Bt, int ldb, int n0,
                                            int k0, int half, int lm){
  const _Float16* p = Bt + (size_t)(n0 + lm) * ldb + k0 + 8 * half;
  V16H r;
  r.h[0] = *(const v8h*)p;
  r.h[1] = *(const v8h*)(p + 16);
  return r.v;
}

__device__ __forceinline__ v16h load_b_frag_g(const _Float16* Bt, int ldb, int n0,
                                              int k0, int half, int lm, int Nn){
  if (n0 + lm >= Nn) return zero_frag();
  return load_b_frag(Bt, ldb, n0, k0, half, lm);
}

// ---------------------------------------------------------------------------
// Generic TN GEMM: C[M,N] = A[M,K](f16,row) * Bt[N,K](f16,row)^T  (+ epilogue)
// block = 128 threads = 4 waves; each wave: 32(M) x 64(N) tile.
// grid = (ceil(N/64), M/128, batches | k-chunks)
// kChunk == 0 : full-K, blockIdx.z selects batch (sA/sB strides).
// kChunk  > 0 : split-K, blockIdx.z selects K-chunk [z*kChunk, (z+1)*kChunk).
// GUARD: compile-time N-edge guarding (only for N % 64 != 0 cases).
// ---------------------------------------------------------------------------
template <bool GUARD, class Epi>
__global__ void __launch_bounds__(128)
gemm_tn(const _Float16* A, int lda, size_t sA,
        const _Float16* Bt, int ldb, size_t sB,
        int Nn, int K, int kChunk, Epi epi)
{
  const int lane = threadIdx.x & 31;
  const int wv   = threadIdx.x >> 5;
  const int half = lane >> 4, lm = lane & 15;
  const int m0 = (blockIdx.y * 4 + wv) * 32;
  const int n0 = blockIdx.x * 64;
  const int bz = blockIdx.z;
  const _Float16* Ab = A + (size_t)bz * sA;
  const _Float16* Bb = Bt + (size_t)bz * sB;
  const int kStart = kChunk ? bz * kChunk : 0;
  const int kEnd   = kChunk ? kStart + kChunk : K;

  v8f acc[2][4] = {};

  // current fragment set (preload first k-step)
  v16h a0c = load_a_frag(Ab, lda, m0,      kStart, half, lm);
  v16h a1c = load_a_frag(Ab, lda, m0 + 16, kStart, half, lm);
  v16h bc0 = GUARD ? load_b_frag_g(Bb, ldb, n0,      kStart, half, lm, Nn)
                   : load_b_frag(Bb, ldb, n0,      kStart, half, lm);
  v16h bc1 = GUARD ? load_b_frag_g(Bb, ldb, n0 + 16, kStart, half, lm, Nn)
                   : load_b_frag(Bb, ldb, n0 + 16, kStart, half, lm);
  v16h bc2 = GUARD ? load_b_frag_g(Bb, ldb, n0 + 32, kStart, half, lm, Nn)
                   : load_b_frag(Bb, ldb, n0 + 32, kStart, half, lm);
  v16h bc3 = GUARD ? load_b_frag_g(Bb, ldb, n0 + 48, kStart, half, lm, Nn)
                   : load_b_frag(Bb, ldb, n0 + 48, kStart, half, lm);

  // steady state: issue next-iteration loads, then consume current fragments
  for (int k0 = kStart; k0 + 32 < kEnd; k0 += 32){
    const int k1 = k0 + 32;
    v16h a0n = load_a_frag(Ab, lda, m0,      k1, half, lm);
    v16h a1n = load_a_frag(Ab, lda, m0 + 16, k1, half, lm);
    v16h bn0 = GUARD ? load_b_frag_g(Bb, ldb, n0,      k1, half, lm, Nn)
                     : load_b_frag(Bb, ldb, n0,      k1, half, lm);
    v16h bn1 = GUARD ? load_b_frag_g(Bb, ldb, n0 + 16, k1, half, lm, Nn)
                     : load_b_frag(Bb, ldb, n0 + 16, k1, half, lm);
    v16h bn2 = GUARD ? load_b_frag_g(Bb, ldb, n0 + 32, k1, half, lm, Nn)
                     : load_b_frag(Bb, ldb, n0 + 32, k1, half, lm);
    v16h bn3 = GUARD ? load_b_frag_g(Bb, ldb, n0 + 48, k1, half, lm, Nn)
                     : load_b_frag(Bb, ldb, n0 + 48, k1, half, lm);
    if (k1 + 32 < kEnd){
      __builtin_prefetch((const void*)(Ab + (size_t)(m0 + lm) * lda + k1 + 32), 0, 1);
      __builtin_prefetch((const void*)(Bb + (size_t)(n0 + lm) * ldb + k1 + 32), 0, 1);
    }
    acc[0][0] = wmma16(a0c, bc0, acc[0][0]);
    acc[1][0] = wmma16(a1c, bc0, acc[1][0]);
    acc[0][1] = wmma16(a0c, bc1, acc[0][1]);
    acc[1][1] = wmma16(a1c, bc1, acc[1][1]);
    acc[0][2] = wmma16(a0c, bc2, acc[0][2]);
    acc[1][2] = wmma16(a1c, bc2, acc[1][2]);
    acc[0][3] = wmma16(a0c, bc3, acc[0][3]);
    acc[1][3] = wmma16(a1c, bc3, acc[1][3]);
    a0c = a0n; a1c = a1n; bc0 = bn0; bc1 = bn1; bc2 = bn2; bc3 = bn3;
  }
  // final k-step
  acc[0][0] = wmma16(a0c, bc0, acc[0][0]);
  acc[1][0] = wmma16(a1c, bc0, acc[1][0]);
  acc[0][1] = wmma16(a0c, bc1, acc[0][1]);
  acc[1][1] = wmma16(a1c, bc1, acc[1][1]);
  acc[0][2] = wmma16(a0c, bc2, acc[0][2]);
  acc[1][2] = wmma16(a1c, bc2, acc[1][2]);
  acc[0][3] = wmma16(a0c, bc3, acc[0][3]);
  acc[1][3] = wmma16(a1c, bc3, acc[1][3]);

  // C layout: VGPR i -> row m0 + 16*s + i + 8*half, col n0 + 16*j + lm
  #pragma unroll
  for (int s = 0; s < 2; ++s){
    #pragma unroll
    for (int j = 0; j < 4; ++j){
      int n = n0 + 16 * j + lm;
      if (!GUARD || n < Nn){
        #pragma unroll
        for (int i = 0; i < 8; ++i)
          epi(bz, m0 + 16 * s + 8 * half + i, n, acc[s][j][i]);
      }
    }
  }
}

// ------------------------------- Epilogues ---------------------------------
struct EpiF32 {               // plain f32 store (DFT output)
  float* C; int ldc;
  __device__ void operator()(int, int m, int n, float v) const {
    C[(size_t)m * ldc + n] = v;
  }
};
struct EpiLeakyF16 {          // v += bias; leaky; f16 store (time1)
  const float* bias; _Float16* C; int ldc;
  __device__ void operator()(int, int m, int n, float v) const {
    C[(size_t)m * ldc + n] = (_Float16)leakyf(v + bias[n]);
  }
};
struct EpiBiasF32 {           // v += bias; f32 store (time2 -> xn)
  const float* bias; float* C; int ldc;
  __device__ void operator()(int, int m, int n, float v) const {
    C[(size_t)m * ldc + n] = v + bias[n];
  }
};
struct EpiOut {               // irfft result + xn + xs*(emb+1) -> out16
  const float* xn; const float* xs; const float* emb; _Float16* out;
  __device__ void operator()(int r, int t, int e, float v) const {
    float o = v + xn[r * CT + t] + xs[r * CT + t] * (emb[e] + 1.0f);
    out[((size_t)r * CT + t) * CE + e] = (_Float16)o;
  }
};
struct EpiPart {              // split-K partial sums (fc1), deterministic
  float* part;                // [CSPLIT][CR][CH]
  __device__ void operator()(int z, int m, int n, float v) const {
    part[((size_t)z * CR + m) * CH + n] = v;
  }
};
struct EpiFc2 {               // + fb2, scatter to d_out[B,P,N] (f32)
  const float* fb2; float* outp;
  __device__ void operator()(int, int m, int p, float v) const {
    int b = m >> 6, n = m & 63;
    outp[((size_t)b * CP + p) * CN + n] = v + fb2[p];
  }
};

// ---------------------------------------------------------------------------
// Frequency MLP GEMM with on-the-fly A:
//   A[m=(r,f)][k] = f16( leaky( X[r,f] * u1[k] + b1[k] ) ),  Bt = w2T[E][H]
//   epilogue: softshrink(v + b2[e]) -> yallT[r][e][off+f]   (f16)
// M = CR*CF = 131584 (multiple of 128), N = CE, K = CH.
// B loads double-buffered; A fragments generated in VALU while B loads fly.
// ---------------------------------------------------------------------------
__global__ void __launch_bounds__(128)
gemm_freq(const float* Xri, const float* u1v, const float* b1v,
          const _Float16* w2T, const float* b2v, _Float16* yallT, int off)
{
  const int lane = threadIdx.x & 31;
  const int wv   = threadIdx.x >> 5;
  const int half = lane >> 4, lm = lane & 15;
  const int m0 = (blockIdx.y * 4 + wv) * 32;
  const int n0 = blockIdx.x * 64;

  float xv[2];
  #pragma unroll
  for (int s = 0; s < 2; ++s){
    int mrow = m0 + 16 * s + lm;
    int rr = mrow / CF;
    int ff = mrow - CF * rr;
    xv[s] = Xri[(size_t)rr * CF2P + off + ff];
  }

  v8f acc[2][4] = {};

  v16h bc0 = load_b_frag(w2T, CH, n0,      0, half, lm);
  v16h bc1 = load_b_frag(w2T, CH, n0 + 16, 0, half, lm);
  v16h bc2 = load_b_frag(w2T, CH, n0 + 32, 0, half, lm);
  v16h bc3 = load_b_frag(w2T, CH, n0 + 48, 0, half, lm);

  for (int k0 = 0; k0 < CH; k0 += 32){
    const int k1 = k0 + 32;
    v16h bn0, bn1, bn2, bn3;
    if (k1 < CH){               // issue next-step loads first
      bn0 = load_b_frag(w2T, CH, n0,      k1, half, lm);
      bn1 = load_b_frag(w2T, CH, n0 + 16, k1, half, lm);
      bn2 = load_b_frag(w2T, CH, n0 + 32, k1, half, lm);
      bn3 = load_b_frag(w2T, CH, n0 + 48, k1, half, lm);
    }
    // generate A fragments (VALU) while loads are in flight
    V16H a[2];
    #pragma unroll
    for (int i = 0; i < 8; ++i){
      int k = k0 + 8 * half + i;
      float u0 = u1v[k],      c0 = b1v[k];
      float u1 = u1v[k + 16], c1 = b1v[k + 16];
      #pragma unroll
      for (int s = 0; s < 2; ++s){
        a[s].e[i]     = (_Float16)leakyf(xv[s] * u0 + c0);
        a[s].e[i + 8] = (_Float16)leakyf(xv[s] * u1 + c1);
      }
    }
    acc[0][0] = wmma16(a[0].v, bc0, acc[0][0]);
    acc[1][0] = wmma16(a[1].v, bc0, acc[1][0]);
    acc[0][1] = wmma16(a[0].v, bc1, acc[0][1]);
    acc[1][1] = wmma16(a[1].v, bc1, acc[1][1]);
    acc[0][2] = wmma16(a[0].v, bc2, acc[0][2]);
    acc[1][2] = wmma16(a[1].v, bc2, acc[1][2]);
    acc[0][3] = wmma16(a[0].v, bc3, acc[0][3]);
    acc[1][3] = wmma16(a[1].v, bc3, acc[1][3]);
    if (k1 < CH){ bc0 = bn0; bc1 = bn1; bc2 = bn2; bc3 = bn3; }
  }

  #pragma unroll
  for (int s = 0; s < 2; ++s){
    #pragma unroll
    for (int j = 0; j < 4; ++j){
      int e = n0 + 16 * j + lm;
      #pragma unroll
      for (int i = 0; i < 8; ++i){
        int m = m0 + 16 * s + 8 * half + i;
        int r = m / CF, f = m - CF * r;
        float v = sshrink(acc[s][j][i] + b2v[e]);
        yallT[((size_t)r * CE + e) * CF2P + off + f] = (_Float16)v;
      }
    }
  }
}

// ------------------------------ Support kernels ----------------------------
__global__ void k_prep(const float* x, float* xs, _Float16* xs16){
  int idx = blockIdx.x * 256 + threadIdx.x;      // CR*CT threads exactly
  int r = idx >> 9, t = idx & 511;
  int b = r >> 6, n = r & 63;
  float v = x[((size_t)b * CT + t) * CN + n];
  xs[idx] = v;
  xs16[idx] = (_Float16)v;
}

__global__ void k_u1(const float* emb, const float* w1, float* u1v){
  int h = threadIdx.x;                           // CH threads
  float s = 0.0f;
  for (int e = 0; e < CE; ++e) s += emb[e] * w1[e * CH + h];
  u1v[h] = s;
}

__global__ void k_convT(const float* src, _Float16* dst, int Kd, int Nd, size_t total){
  size_t i = (size_t)blockIdx.x * 256 + threadIdx.x;
  if (i >= total) return;
  size_t k = i / (size_t)Nd, n = i - k * (size_t)Nd;
  dst[n * (size_t)Kd + k] = (_Float16)src[i];
}

__global__ void k_fill_u32(unsigned* p, size_t n){
  size_t i = (size_t)blockIdx.x * 256 + threadIdx.x;
  size_t stride = (size_t)gridDim.x * 256;
  for (; i < n; i += stride) p[i] = 0u;
}

// fc1 split-K reduction: act2 = f16(leaky(sum_z part[z] + fb1))
__global__ void k_reduce_fc1(const float* part, const float* fb1, _Float16* act2){
  int idx = blockIdx.x * 256 + threadIdx.x;      // CR*CH threads exactly
  float s = 0.0f;
  for (int z = 0; z < CSPLIT; ++z) s += part[(size_t)z * CR * CH + idx];
  int h = idx & (CH - 1);
  act2[idx] = (_Float16)leakyf(s + fb1[h]);
}

// DFT basis, stored as Bt[j][t] (ld = CT):
//   j <  CF : (1/sqrt(T)) * cos(2*pi*j*t/T)
//   j <  CF2: -(1/sqrt(T)) * sin(2*pi*(j-CF)*t/T)
//   else    : 0  (padding)
__global__ void k_dft_basis(_Float16* W){
  int idx = blockIdx.x * 256 + threadIdx.x;
  if (idx >= CF2P * CT) return;
  int j = idx / CT, t = idx - j * CT;
  const float c = 0.04419417382415922f;          // 1/sqrt(512)
  const float w = 6.283185307179586f / (float)CT;
  float v = 0.0f;
  if (j < CF){
    int ph = (j * t) & (CT - 1);
    v = c * cosf(w * (float)ph);
  } else if (j < CF2){
    int ph = ((j - CF) * t) & (CT - 1);
    v = -c * sinf(w * (float)ph);
  }
  W[idx] = (_Float16)v;
}

// irfft (ortho) basis, stored as A[t][j] (ld = CF2P):
//   xt[t] = (1/sqrt(T)) * [ yr0 + 2*sum_{1..255}(yr cos - yi sin) + yr256*(-1)^t ]
// (imag DC/Nyquist drop out since sin(pi*t)=sin(0)=0)
__global__ void k_irfft_basis(_Float16* W){
  int idx = blockIdx.x * 256 + threadIdx.x;
  if (idx >= CT * CF2P) return;
  int t = idx / CF2P, j = idx - t * CF2P;
  const float c = 0.04419417382415922f;
  const float w = 6.283185307179586f / (float)CT;
  float v = 0.0f;
  if (j < CF){
    int f = j;
    float mfac = (f == 0 || f == CF - 1) ? 1.0f : 2.0f;
    v = c * mfac * cosf(w * (float)((f * t) & (CT - 1)));
  } else if (j < CF2){
    int f = j - CF;
    float mfac = (f == 0 || f == CF - 1) ? 1.0f : 2.0f;
    v = -c * mfac * sinf(w * (float)((f * t) & (CT - 1)));
  }
  W[idx] = (_Float16)v;
}

// ---------------------------------------------------------------------------
extern "C" void kernel_launch(void* const* d_in, const int* in_sizes, int n_in,
                              void* d_out, int out_size, void* d_ws, size_t ws_size,
                              hipStream_t stream)
{
  (void)in_sizes; (void)n_in; (void)out_size; (void)ws_size;
  const float* x   = (const float*)d_in[0];
  const float* emb = (const float*)d_in[1];
  const float* w1  = (const float*)d_in[2];
  const float* b1  = (const float*)d_in[3];
  const float* w2  = (const float*)d_in[4];
  const float* b2  = (const float*)d_in[5];
  const float* tw1 = (const float*)d_in[6];
  const float* tb1 = (const float*)d_in[7];
  const float* tw2 = (const float*)d_in[8];
  const float* tb2 = (const float*)d_in[9];
  const float* fw1 = (const float*)d_in[10];
  const float* fb1 = (const float*)d_in[11];
  const float* fw2 = (const float*)d_in[12];
  const float* fb2 = (const float*)d_in[13];
  float* outp = (float*)d_out;

  // Workspace carve-out (all offsets 256B aligned). Total ~195 MB.
  char* ws = (char*)d_ws;
  size_t off = 0;
  auto carve = [&](size_t bytes) -> void* {
    void* p = ws + off;
    off += (bytes + 255) & ~(size_t)255;
    return p;
  };
  float*    xs    = (float*)   carve((size_t)CR * CT * 4);
  _Float16* xs16  = (_Float16*)carve((size_t)CR * CT * 2);
  _Float16* WdftT = (_Float16*)carve((size_t)CF2P * CT * 2);
  float*    Xri   = (float*)   carve((size_t)CR * CF2P * 4);
  float*    u1v   = (float*)   carve((size_t)CH * 4);
  _Float16* w2T   = (_Float16*)carve((size_t)CE * CH * 2);
  _Float16* tw1T  = (_Float16*)carve((size_t)CH * CT * 2);
  _Float16* tw2T  = (_Float16*)carve((size_t)CT * CH * 2);
  _Float16* fw2T  = (_Float16*)carve((size_t)CP * CH * 2);
  _Float16* actT  = (_Float16*)carve((size_t)CR * CH * 2);
  _Float16* act2  = (_Float16*)carve((size_t)CR * CH * 2);
  float*    xnv   = (float*)   carve((size_t)CR * CT * 4);
  _Float16* Wir   = (_Float16*)carve((size_t)CT * CF2P * 2);
  float*    hpart = (float*)   carve((size_t)CSPLIT * CR * CH * 4); // 16.8 MB
  _Float16* fw1T  = (_Float16*)carve((size_t)CT * CE * CH * 2);     // 33.5 MB
  _Float16* yallT = (_Float16*)carve((size_t)CR * CE * CF2P * 2);   // 71.3 MB
  _Float16* out16 = (_Float16*)carve((size_t)CR * CT * CE * 2);     // 67.1 MB

  const dim3 blk256(256), blk128(128);

  // Stage 0: transpose x -> xs / xs16 ; u1 = emb @ w1 ; weight transposes (f16)
  k_prep<<<dim3(CR * CT / 256), blk256, 0, stream>>>(x, xs, xs16);
  k_u1  <<<dim3(1), dim3(CH), 0, stream>>>(emb, w1, u1v);
  k_convT<<<dim3((CH * CE + 255) / 256), blk256, 0, stream>>>(w2,  w2T,  CH, CE, (size_t)CH * CE);
  k_convT<<<dim3((CT * CH + 255) / 256), blk256, 0, stream>>>(tw1, tw1T, CT, CH, (size_t)CT * CH);
  k_convT<<<dim3((CH * CT + 255) / 256), blk256, 0, stream>>>(tw2, tw2T, CH, CT, (size_t)CH * CT);
  k_convT<<<dim3(((size_t)CT * CE * CH + 255) / 256), blk256, 0, stream>>>(fw1, fw1T, CT * CE, CH, (size_t)CT * CE * CH);
  k_convT<<<dim3((CH * CP + 255) / 256), blk256, 0, stream>>>(fw2, fw2T, CH, CP, (size_t)CH * CP);

  // Stage 1: ortho DFT as GEMM: Xri[CR][CF2P] = xs16 @ WdftT^T   (N=544 -> guarded)
  k_dft_basis<<<dim3((CF2P * CT + 255) / 256), blk256, 0, stream>>>(WdftT);
  gemm_tn<true, EpiF32><<<dim3((CF2P + 63) / 64, CR / 128, 1), blk128, 0, stream>>>(
      xs16, CT, 0, WdftT, CT, 0, CF2P, CT, 0, EpiF32{Xri, CF2P});

  // Stage 2: frequency MLP (layer1 fused in-register, layer2 = WMMA GEMM)
  k_fill_u32<<<dim3(2048), blk256, 0, stream>>>((unsigned*)yallT,
                                                (size_t)CR * CE * CF2P / 2);
  gemm_freq<<<dim3(CE / 64, (CR * CF) / 128, 1), blk128, 0, stream>>>(
      Xri, u1v, b1, w2T, b2, yallT, 0);    // real part
  gemm_freq<<<dim3(CE / 64, (CR * CF) / 128, 1), blk128, 0, stream>>>(
      Xri, u1v, b1, w2T, b2, yallT, CF);   // imag part

  // Stage 3: time-domain MLP: xn = leaky(xs@tw1+tb1)@tw2 + tb2
  gemm_tn<false, EpiLeakyF16><<<dim3(CH / 64, CR / 128, 1), blk128, 0, stream>>>(
      xs16, CT, 0, tw1T, CT, 0, CH, CT, 0, EpiLeakyF16{tb1, actT, CH});
  gemm_tn<false, EpiBiasF32><<<dim3(CT / 64, CR / 128, 1), blk128, 0, stream>>>(
      actT, CH, 0, tw2T, CH, 0, CT, CH, 0, EpiBiasF32{tb2, xnv, CT});

  // Stage 4: batched irfft GEMM + fused (xn + xs*(emb+1)) epilogue -> out16
  k_irfft_basis<<<dim3((CT * CF2P + 255) / 256), blk256, 0, stream>>>(Wir);
  gemm_tn<false, EpiOut><<<dim3(CE / 64, CT / 128, CR), blk128, 0, stream>>>(
      Wir, CF2P, 0, yallT, CF2P, (size_t)CE * CF2P, CE, CF2P, 0,
      EpiOut{xnv, xs, emb, out16});

  // Stage 5: fc1 with deterministic split-K (K = 65536 -> 32 chunks of 2048)
  gemm_tn<false, EpiPart><<<dim3(CH / 64, CR / 128, CSPLIT), blk128, 0, stream>>>(
      out16, CT * CE, 0, fw1T, CT * CE, 0, CH, CT * CE, (CT * CE) / CSPLIT,
      EpiPart{hpart});
  k_reduce_fc1<<<dim3(CR * CH / 256), blk256, 0, stream>>>(hpart, fb1, act2);

  // Stage 6: fc2 + transpose to [B,P,N]   (N=96 -> guarded)
  gemm_tn<true, EpiFc2><<<dim3((CP + 63) / 64, CR / 128, 1), blk128, 0, stream>>>(
      act2, CH, 0, fw2T, CH, 0, CP, CH, 0, EpiFc2{fb2, outp});
}
